// AGCN_60224031424871
// MI455X (gfx1250) — matrix operand order repaired and checked by reading, not compiled, not run on hardware
//
#include <hip/hip_runtime.h>
#include <hip/hip_bf16.h>
#include <math.h>

// ---------------------------------------------------------------------------
// Problem constants (match reference)
// ---------------------------------------------------------------------------
#define NN   20000
#define EE   160000
#define GG   64
#define DESM 1280
#define DD   512
#define OUTD 500

typedef __bf16 bf16;
typedef __attribute__((ext_vector_type(16))) __bf16 v16bf;
typedef __attribute__((ext_vector_type(8)))  float  v8f;

// f32 -> bf16: native cast, compiler emits hardware RNE convert (v_cvt_*bf16*)
__device__ __forceinline__ bf16 f2bf(float f) { return (bf16)f; }

// ---------------------------------------------------------------------------
// WMMA GEMM:  C[M,512] = A[M,K] @ W[K,512] (+bias)   (K % 32 == 0)
// Optional row gather on A (embedding lookup).
// Block: 256 threads = 8 waves. Block tile: 64 rows (4 M-tiles) x 512 cols.
// Wave w owns columns [w*64, w*64+64) = 4 WMMA N-tiles; B fragments are
// loaded once per K-step and reused across the 4 M-tiles (16 WMMAs/step).
// ---------------------------------------------------------------------------
__global__ __launch_bounds__(256)
void gemm_nc512_bf16(const float* __restrict__ A,
                     const int*   __restrict__ gather,
                     const float* __restrict__ W,
                     const float* __restrict__ bias,
                     float* __restrict__ C,
                     int M, int K)
{
    __shared__ bf16 lA[64 * 32];     // 4 M-tiles of 16x32 bf16

    const int tid  = threadIdx.x;
    const int wave = tid >> 5;
    const int lane = tid & 31;
    const int mb0  = blockIdx.x * 64;

    const int row  = lane & 15;      // M-row (A) / N-col (B,C) within tile
    const int half = lane >> 4;      // 0: lanes 0-15, 1: lanes 16-31
    const int colBase = wave * 64 + row;

    v8f acc[4][4];
#pragma unroll
    for (int mt = 0; mt < 4; ++mt)
#pragma unroll
        for (int t = 0; t < 4; ++t) acc[mt][t] = (v8f){};

    for (int k0 = 0; k0 < K; k0 += 32) {
        // ---- stage A tile (64x32) into LDS as bf16 ----
        for (int i = tid; i < 64 * 32; i += 256) {
            int r  = i >> 5;
            int kk = i & 31;
            int gm = mb0 + r;
            float v = 0.f;
            if (gm < M) {
                long srcRow = gather ? (long)gather[gm] : (long)gm;
                const float* ap = A + srcRow * K + k0 + kk;
                v = *ap;
                if (k0 + 32 < K) __builtin_prefetch(ap + 32, 0, 3);
            }
            lA[i] = f2bf(v);
        }
        __syncthreads();

        // ---- A fragments (contiguous K order, matching B) ----
        // lane<16:  K = 0..15 in v0-7
        // lane>=16: K = 16..31 in v0-7; 2 K per VGPR
        v16bf af[4];
#pragma unroll
        for (int mt = 0; mt < 4; ++mt) {
            const bf16* base = lA + (mt * 16 + row) * 32;
#pragma unroll
            for (int v = 0; v < 8; ++v) {
                int kb = half * 16 + v * 2;
                af[mt][2 * v]     = base[kb];
                af[mt][2 * v + 1] = base[kb + 1];
            }
        }

        // ---- B fragments (32x16, col = lane%16; lane<16 -> K 0..15,
        //      lane>=16 -> K 16..31, sequential 2 per VGPR) ----
        const float* Wk = W + (long)(k0 + half * 16) * 512;
        v16bf b0, b1, b2, b3;
#pragma unroll
        for (int j = 0; j < 16; ++j) {
            const float* wr = Wk + (long)j * 512 + colBase;
            b0[j] = f2bf(wr[0]);
            b1[j] = f2bf(wr[16]);
            b2[j] = f2bf(wr[32]);
            b3[j] = f2bf(wr[48]);
        }

#pragma unroll
        for (int mt = 0; mt < 4; ++mt) {
            acc[mt][0] = __builtin_amdgcn_wmma_f32_16x16x32_bf16(
                false, af[mt], false, b0, (short)0, acc[mt][0], false, false);
            acc[mt][1] = __builtin_amdgcn_wmma_f32_16x16x32_bf16(
                false, af[mt], false, b1, (short)0, acc[mt][1], false, false);
            acc[mt][2] = __builtin_amdgcn_wmma_f32_16x16x32_bf16(
                false, af[mt], false, b2, (short)0, acc[mt][2], false, false);
            acc[mt][3] = __builtin_amdgcn_wmma_f32_16x16x32_bf16(
                false, af[mt], false, b3, (short)0, acc[mt][3], false, false);
        }

        __syncthreads();
    }

    // ---- store C (f32 C/D layout: VGPR r -> row r + 8*half, col = lane%16) ----
#pragma unroll
    for (int mt = 0; mt < 4; ++mt) {
#pragma unroll
        for (int t = 0; t < 4; ++t) {
            int n = wave * 64 + t * 16 + row;
            float badd = bias ? bias[n] : 0.f;
#pragma unroll
            for (int r = 0; r < 8; ++r) {
                int m = mb0 + mt * 16 + half * 8 + r;
                if (m < M) C[(long)m * 512 + n] = acc[mt][t][r] + badd;
            }
        }
    }
}

// ---------------------------------------------------------------------------
// Elementwise / graph kernels
// ---------------------------------------------------------------------------
__global__ void fill_f32(float* __restrict__ p, float v, long n)
{
    long i = (long)blockIdx.x * 256 + threadIdx.x;
    if (i < n) p[i] = v;
}

// h = relu(esm + aa); xe = relu(esm)
__global__ void add_relu2(const float* __restrict__ esm, const float* __restrict__ aa,
                          float* __restrict__ h, float* __restrict__ xe, long n)
{
    long i = (long)blockIdx.x * 256 + threadIdx.x;
    if (i >= n) return;
    float e = esm[i];
    h[i]  = fmaxf(e + aa[i], 0.f);
    xe[i] = fmaxf(e, 0.f);
}

__global__ void edge_deg(const int* __restrict__ dst, float* __restrict__ deg, int E)
{
    int e = blockIdx.x * 256 + threadIdx.x;
    if (e < E) atomicAdd(&deg[dst[e]], 1.0f);
}

// in: deg (without self loop); out: dis = rsqrt(deg+1) in-place, selfn = 1/(deg+1)
__global__ void node_norm(float* __restrict__ dis, float* __restrict__ selfn, int N)
{
    int n = blockIdx.x * 256 + threadIdx.x;
    if (n >= N) return;
    float d = dis[n] + 1.0f;
    dis[n]   = rsqrtf(d);
    selfn[n] = 1.0f / d;
}

__global__ void edge_norm(const int* __restrict__ src, const int* __restrict__ dst,
                          const float* __restrict__ dis, float* __restrict__ enorm, int E)
{
    int e = blockIdx.x * 256 + threadIdx.x;
    if (e < E) enorm[e] = dis[src[e]] * dis[dst[e]];
}

// agg[dst] += hW[src] * enorm   (128 threads per edge, float4 each)
__global__ __launch_bounds__(256)
void edge_scatter(const int* __restrict__ src, const int* __restrict__ dst,
                  const float* __restrict__ enorm,
                  const float* __restrict__ hW, float* __restrict__ agg, int E)
{
    long gid = (long)blockIdx.x * 256 + threadIdx.x;
    long e = gid >> 7;
    if (e >= E) return;
    int c = (int)(gid & 127) * 4;
    int s = src[e], t = dst[e];
    float w = enorm[e];
    const float4 v = *(const float4*)(hW + (long)s * 512 + c);
    float* a = agg + (long)t * 512 + c;
    atomicAdd(a + 0, v.x * w);
    atomicAdd(a + 1, v.y * w);
    atomicAdd(a + 2, v.z * w);
    atomicAdd(a + 3, v.w * w);
}

// out = relu(agg + hW*selfn + b); feat = first ? out : feat + out
__global__ void gcn_combine(float* __restrict__ feat, const float* __restrict__ hW,
                            const float* __restrict__ agg, const float* __restrict__ selfn,
                            const float* __restrict__ bias, int N, int first)
{
    long i = (long)blockIdx.x * 256 + threadIdx.x;
    if (i >= (long)N * 512) return;
    int n = (int)(i >> 9);
    int d = (int)(i & 511);
    float v = agg[i] + hW[i] * selfn[n] + bias[d];
    v = fmaxf(v, 0.f);
    feat[i] = first ? v : feat[i] + v;
}

// per-graph max pool; feat >= 0 so uint-bit atomicMax with 0-init is exact
__global__ void seg_max(const float* __restrict__ feat, const int* __restrict__ batch,
                        float* __restrict__ g, int N)
{
    long i = (long)blockIdx.x * 256 + threadIdx.x;
    if (i >= (long)N * 512) return;
    int n = (int)(i >> 9);
    int d = (int)(i & 511);
    atomicMax((unsigned int*)&g[(long)batch[n] * 512 + d], __float_as_uint(feat[i]));
}

__global__ void pool_mix(const float* __restrict__ g1, const float* __restrict__ g3,
                         const float* __restrict__ w1, float* __restrict__ g)
{
    int i = blockIdx.x * 256 + threadIdx.x;
    if (i < GG * 512) g[i] = w1[0] * g1[i] + w1[1] * g3[i];
}

// t[64,1024] = relu(g[64,512] @ w[512,1024] + b)
__global__ void readout1(const float* __restrict__ g, const float* __restrict__ w,
                         const float* __restrict__ b, float* __restrict__ t)
{
    int i = blockIdx.y;
    int j = blockIdx.x * 256 + threadIdx.x;
    if (j >= 1024) return;
    float acc = b[j];
    const float* gr = g + (long)i * 512;
    for (int k = 0; k < 512; ++k) acc += gr[k] * w[(long)k * 1024 + j];
    t[(long)i * 1024 + j] = fmaxf(acc, 0.f);
}

// y[64,500] = sigmoid(t[64,1024] @ w[1024,500] + b)
__global__ void readout2(const float* __restrict__ t, const float* __restrict__ w,
                         const float* __restrict__ b, float* __restrict__ y)
{
    int i = blockIdx.y;
    int j = blockIdx.x * 256 + threadIdx.x;
    if (j >= OUTD) return;
    float acc = b[j];
    const float* tr = t + (long)i * 1024;
    for (int k = 0; k < 1024; ++k) acc += tr[k] * w[(long)k * OUTD + j];
    y[(long)i * OUTD + j] = 1.f / (1.f + expf(-acc));
}

// ---------------------------------------------------------------------------
// Launch
// ---------------------------------------------------------------------------
extern "C" void kernel_launch(void* const* d_in, const int* in_sizes, int n_in,
                              void* d_out, int out_size, void* d_ws, size_t ws_size,
                              hipStream_t stream)
{
    const int*   native_x   = (const int*)  d_in[0];
    const float* x          = (const float*)d_in[1];
    const int*   edge_index = (const int*)  d_in[2];
    const int*   batch      = (const int*)  d_in[3];
    const float* embed_tbl  = (const float*)d_in[4];
    const float* proj_aa_w  = (const float*)d_in[5];
    const float* proj_aa_b  = (const float*)d_in[6];
    const float* proj_esm_w = (const float*)d_in[7];
    const float* proj_esm_b = (const float*)d_in[8];
    const float* gcn_w      = (const float*)d_in[9];
    const float* gcn_b      = (const float*)d_in[10];
    const float* ro1_w      = (const float*)d_in[11];
    const float* ro1_b      = (const float*)d_in[12];
    const float* ro2_w      = (const float*)d_in[13];
    const float* ro2_b      = (const float*)d_in[14];
    const float* weight1    = (const float*)d_in[15];

    const int* srcp = edge_index;       // edge_index[0]
    const int* dstp = edge_index + EE;  // edge_index[1]

    const long ND = (long)NN * 512;

    // workspace layout (floats)
    float* ws    = (float*)d_ws;
    float* bufA  = ws;                  // x_esm_lin -> hW
    float* bufB  = bufA + ND;           // x_aa      -> agg
    float* bufH  = bufB + ND;           // feat stack 1
    float* bufE  = bufH + ND;           // feat stack 2
    float* dis   = bufE + ND;           // deg -> dis   [N]
    float* selfn = dis + NN;            // [N]
    float* enorm = selfn + NN;          // [E]
    float* g1    = enorm + EE;          // [64*512]
    float* g3    = g1 + GG * 512;       // [64*512]
    float* gmix  = g3 + GG * 512;       // [64*512]
    float* t1    = gmix + GG * 512;     // [64*1024]

    const int BLK = 256;
    const int mblocks = (NN + 63) / 64;

    // 1) x_esm_lin = x @ proj_esm_w + b
    gemm_nc512_bf16<<<mblocks, BLK, 0, stream>>>(x, nullptr, proj_esm_w, proj_esm_b,
                                                 bufA, NN, DESM);
    // 2) x_aa = embed[native_x] @ proj_aa_w + b
    gemm_nc512_bf16<<<mblocks, BLK, 0, stream>>>(embed_tbl, native_x, proj_aa_w, proj_aa_b,
                                                 bufB, NN, 96);
    // 3) h = relu(lin + aa); x_esm = relu(lin)
    add_relu2<<<(int)((ND + BLK - 1) / BLK), BLK, 0, stream>>>(bufA, bufB, bufH, bufE, ND);

    // 4) degree / normalization
    fill_f32<<<(NN + BLK - 1) / BLK, BLK, 0, stream>>>(dis, 0.f, NN);
    edge_deg<<<(EE + BLK - 1) / BLK, BLK, 0, stream>>>(dstp, dis, EE);
    node_norm<<<(NN + BLK - 1) / BLK, BLK, 0, stream>>>(dis, selfn, NN);
    edge_norm<<<(EE + BLK - 1) / BLK, BLK, 0, stream>>>(srcp, dstp, dis, enorm, EE);

    // 5) two GCN stacks, 3 layers each
    const long scatterThreads = (long)EE * 128;
    for (int s = 0; s < 2; ++s) {
        float* feat = s ? bufE : bufH;
        for (int l = 0; l < 3; ++l) {
            gemm_nc512_bf16<<<mblocks, BLK, 0, stream>>>(feat, nullptr,
                                                         gcn_w + (long)l * 512 * 512,
                                                         nullptr, bufA, NN, 512);
            fill_f32<<<(int)((ND + BLK - 1) / BLK), BLK, 0, stream>>>(bufB, 0.f, ND);
            edge_scatter<<<(int)((scatterThreads + BLK - 1) / BLK), BLK, 0, stream>>>(
                srcp, dstp, enorm, bufA, bufB, EE);
            gcn_combine<<<(int)((ND + BLK - 1) / BLK), BLK, 0, stream>>>(
                feat, bufA, bufB, selfn, gcn_b + (long)l * 512, NN, l == 0);
        }
    }

    // 6) per-graph max pool
    fill_f32<<<(GG * 512 + BLK - 1) / BLK, BLK, 0, stream>>>(g1, 0.f, GG * 512);
    fill_f32<<<(GG * 512 + BLK - 1) / BLK, BLK, 0, stream>>>(g3, 0.f, GG * 512);
    seg_max<<<(int)((ND + BLK - 1) / BLK), BLK, 0, stream>>>(bufH, batch, g1, NN);
    seg_max<<<(int)((ND + BLK - 1) / BLK), BLK, 0, stream>>>(bufE, batch, g3, NN);

    // 7) mix + readout
    pool_mix<<<(GG * 512 + BLK - 1) / BLK, BLK, 0, stream>>>(g1, g3, weight1, gmix);
    readout1<<<dim3(4, GG), BLK, 0, stream>>>(gmix, ro1_w, ro1_b, t1);
    readout2<<<dim3(2, GG), BLK, 0, stream>>>(t1, ro2_w, ro2_b, (float*)d_out);
}
